// NR_GraphAttentionCross_52853867545024
// MI455X (gfx1250) — matrix operand order repaired and checked by reading, not compiled, not run on hardware
//
#include <hip/hip_runtime.h>

// ---------------------------------------------------------------------------
// NR_GraphAttentionCross for MI455X (gfx1250, wave32, WMMA bf16 + TDM)
//
// Pipeline (layer-invariant parts computed once):
//   k_tanh      : feats = tanh(features); write out[:,0:128]
//   k_relnorm   : rel_norm = normalize(rel_emb); rel_logit[l] = rel_norm@ak_l
//   k_mu        : column sums of concat_fea (for Gram centering)
//   k_buildC    : Cbf[E,256] bf16 = [fc[adj0] | fc[adj1]]
//   k_gram      : G = C^T C via v_wmma_f32_16x16x32_bf16 (syrk, atomic accum)
//   k_gfix      : G -= mu mu^T / E   (centered Gram)
//   k_jacobi    : parallel cyclic Jacobi eigensolver, 1 WG, 10 sweeps
//   k_prepV     : sort eigenpairs desc, Vt[n][k] bf16 = V[k][col_n]*(sqrt(w)+1e-5)^-1/2
//   k_proj      : proj[E,128] f32 = C @ Vscaled via WMMA; A-panel staged into
//                 LDS by the Tensor Data Mover (tensor_load_to_lds +
//                 s_wait_tensorcnt), consumed via ds_load_b128.
//   k_projnorm  : inf->0, row-normalize, logits vs ake0/ake1
//   k_elog      : per-edge rel logits = sign(r_val)*rel_logit[rel]
//   k_rowptr    : CSR row offsets from sorted adj0 (binary search)
//   k_agg (x2)  : per-node (wave-per-node) segment softmax + Householder
//                 reflections + weighted segment sum + tanh; writes out blocks
//
// Workspace requirement: ~460 MB (dominated by Cbf 204.8MB + proj 204.8MB).
// ---------------------------------------------------------------------------

#define N_   40000
#define D_   128
#define R_   1000
#define E_   400000
#define TD_  256           // 2*D

typedef __attribute__((ext_vector_type(16))) __bf16 v16bf;
typedef __attribute__((ext_vector_type(8)))  __bf16 v8bf;
typedef __attribute__((ext_vector_type(8)))  float  v8f;
typedef __attribute__((ext_vector_type(4)))  unsigned int u32x4;
typedef __attribute__((ext_vector_type(8)))  int i32x8;
typedef __attribute__((ext_vector_type(4)))  int i32x4;

// ------------------------- small helpers -----------------------------------

__device__ __forceinline__ float blockReduceSum128(float v, float* red, int d) {
  red[d] = v; __syncthreads();
#pragma unroll
  for (int s = 64; s > 0; s >>= 1) { if (d < s) red[d] += red[d + s]; __syncthreads(); }
  float r = red[0]; __syncthreads();
  return r;
}

__global__ void k_zero(float* __restrict__ p, int n) {
  int i = blockIdx.x * 256 + threadIdx.x;
  if (i < n) p[i] = 0.f;
}

// ------------------------- phase A kernels ----------------------------------

__global__ void k_tanh(const float* __restrict__ f, float* __restrict__ feats,
                       float* __restrict__ out) {
  int i = blockIdx.x * 256 + threadIdx.x;
  if (i >= N_ * D_) return;
  float v = tanhf(f[i]);
  feats[i] = v;
  out[(size_t)(i >> 7) * 384 + (i & 127)] = v;
}

__global__ void k_relnorm(const float* __restrict__ rel_emb,
                          const float* __restrict__ ak0, const float* __restrict__ ak1,
                          float* __restrict__ rel_nrm, float* __restrict__ rel_lg) {
  __shared__ float red[128];
  int r = blockIdx.x, d = threadIdx.x;
  float v = rel_emb[(size_t)r * D_ + d];
  float ss = blockReduceSum128(v * v, red, d);
  float vn = v / fmaxf(sqrtf(ss), 1e-12f);
  rel_nrm[(size_t)r * D_ + d] = vn;
  float l0 = blockReduceSum128(vn * ak0[d], red, d);
  if (d == 0) rel_lg[r] = l0;
  float l1 = blockReduceSum128(vn * ak1[d], red, d);
  if (d == 0) rel_lg[R_ + r] = l1;
}

__global__ void k_mu(const int* __restrict__ adj0, const int* __restrict__ adj1,
                     const float* __restrict__ fc, float* __restrict__ mu) {
  int c = threadIdx.x;                        // component 0..255
  long e0 = (long)blockIdx.x * 1024;
  long e1 = e0 + 1024; if (e1 > E_) e1 = E_;
  float acc = 0.f;
  int d = c & 127;
  for (long e = e0; e < e1; ++e) {
    int node = (c < 128) ? adj0[e] : adj1[e];
    acc += fc[(size_t)node * D_ + d];
  }
  atomicAdd(&mu[c], acc);
}

__global__ void k_buildC(const int* __restrict__ adj0, const int* __restrict__ adj1,
                         const float* __restrict__ fc, __bf16* __restrict__ Cbf) {
  int e = blockIdx.x, c = threadIdx.x;
  int node = (c < 128) ? adj0[e] : adj1[e];
  Cbf[(size_t)e * TD_ + c] = (__bf16)fc[(size_t)node * D_ + (c & 127)];
}

// G = C^T C. grid: x = 256 output tiles (16x16 each), y = edge chunks of 4096.
// One wave per workgroup; f32 accumulation in WMMA, atomic reduce into G.
__global__ void __launch_bounds__(32)
k_gram(const __bf16* __restrict__ Cbf, float* __restrict__ G) {
  const int lane = threadIdx.x;
  const int hi = lane >> 4, lm = lane & 15;
  const int ti = (blockIdx.x >> 4) << 4;
  const int tj = (blockIdx.x & 15) << 4;
  long e0b = (long)blockIdx.y * 4096;
  long e1 = e0b + 4096; if (e1 > E_) e1 = E_;   // E_ and chunk are multiples of 32
  v8f acc = {};
  for (long e0 = e0b; e0 < e1; e0 += 32) {
    v16bf a, b;
#pragma unroll
    for (int i = 0; i < 16; ++i) {
      long ea = e0 + ((i & 7) + 8 * hi + 16 * (i >> 3));     // A[M=ti+lm][K=e]
      a[i] = Cbf[(size_t)ea * TD_ + ti + lm];
      long eb = e0 + ((i & 7) + 8 * hi + 16 * (i >> 3));     // B[K=e][N=tj+lm]
      b[i] = Cbf[(size_t)eb * TD_ + tj + lm];
    }
    acc = __builtin_amdgcn_wmma_f32_16x16x32_bf16(false, a, false, b,
                                                  (short)0, acc, false, false);
  }
#pragma unroll
  for (int v = 0; v < 8; ++v) {
    int gi = ti + v + 8 * hi, gj = tj + lm;
    atomicAdd(&G[gi * TD_ + gj], acc[v]);
  }
}

__global__ void k_gfix(float* __restrict__ G, const float* __restrict__ mu) {
  int idx = blockIdx.x * 256 + threadIdx.x;
  if (idx >= TD_ * TD_) return;
  int i = idx >> 8, j = idx & 255;
  G[idx] -= mu[i] * mu[j] * (1.0f / (float)E_);
}

__global__ void k_initV(float* __restrict__ V) {
  int idx = blockIdx.x * 256 + threadIdx.x;
  if (idx >= TD_ * TD_) return;
  V[idx] = ((idx >> 8) == (idx & 255)) ? 1.f : 0.f;
}

// Parallel cyclic Jacobi: 128 disjoint rotations per round, 255 rounds/sweep.
__global__ void k_jacobi(float* __restrict__ G, float* __restrict__ V) {
  __shared__ float cs[128], sn[128];
  __shared__ int pp[128], qq[128];
  __shared__ int perm[256];
  int t = threadIdx.x;
  perm[t] = t;
  __syncthreads();
  for (int sweep = 0; sweep < 10; ++sweep) {
    for (int round = 0; round < 255; ++round) {
      if (t < 128) {
        int p = perm[t], q = perm[255 - t];
        if (p > q) { int x = p; p = q; q = x; }
        pp[t] = p; qq[t] = q;
        float app = G[p * TD_ + p], aqq = G[q * TD_ + q], apq = G[p * TD_ + q];
        float c = 1.f, s = 0.f;
        if (fabsf(apq) > 1e-24f) {
          float tau = (aqq - app) / (2.f * apq);
          float tt = ((tau >= 0.f) ? 1.f : -1.f) / (fabsf(tau) + sqrtf(1.f + tau * tau));
          c = 1.f / sqrtf(1.f + tt * tt);
          s = tt * c;
        }
        cs[t] = c; sn[t] = s;
      }
      __syncthreads();
      // row pass: A1 = J^T G  (disjoint row pairs)
      for (int w = t; w < 128 * TD_; w += 256) {
        int m = w >> 8, k = w & 255;
        int p = pp[m], q = qq[m];
        float c = cs[m], s = sn[m];
        float gp = G[p * TD_ + k], gq = G[q * TD_ + k];
        G[p * TD_ + k] = c * gp - s * gq;
        G[q * TD_ + k] = s * gp + c * gq;
      }
      __syncthreads();
      // col pass: A2 = A1 J,  V = V J (disjoint column pairs)
      for (int w = t; w < 128 * TD_; w += 256) {
        int m = w >> 8, k = w & 255;
        int p = pp[m], q = qq[m];
        float c = cs[m], s = sn[m];
        float gp = G[k * TD_ + p], gq = G[k * TD_ + q];
        G[k * TD_ + p] = c * gp - s * gq;
        G[k * TD_ + q] = s * gp + c * gq;
        float vp = V[k * TD_ + p], vq = V[k * TD_ + q];
        V[k * TD_ + p] = c * vp - s * vq;
        V[k * TD_ + q] = s * vp + c * vq;
      }
      __syncthreads();
      int val = perm[t];
      __syncthreads();
      if (t >= 2) perm[t - 1] = val;
      else if (t == 1) perm[255] = val;
      __syncthreads();
    }
  }
}

// Sort eigenpairs descending, build Vt[n][k] = V[k][col_n] * (sqrt(w)+1e-5)^-1/2
__global__ void k_prepV(const float* __restrict__ G, const float* __restrict__ V,
                        __bf16* __restrict__ Vt) {
  __shared__ float w[256];
  __shared__ int idx[256];
  int t = threadIdx.x;
  w[t] = G[t * TD_ + t];
  __syncthreads();
  float wt = w[t];
  int r = 0;
  for (int j = 0; j < 256; ++j) {
    float wj = w[j];
    if (wj > wt || (wj == wt && j < t)) ++r;
  }
  idx[r] = t;                       // descending rank r -> original column t
  __syncthreads();
  for (int z = t; z < D_ * TD_; z += 256) {
    int d = z >> 8, k = z & 255;
    int col = idx[d];
    float sv = sqrtf(fmaxf(w[col], 0.f));
    float scale = 1.f / sqrtf(sv + 1e-5f);
    Vt[(size_t)d * TD_ + k] = (__bf16)(V[k * TD_ + col] * scale);
  }
}

// proj[E,128] = C @ Vscaled. 8 waves/WG, wave w -> output cols [16w,16w+16),
// block -> 16 edge rows. The [16x256] bf16 A-panel is DMA'd into LDS by the
// Tensor Data Mover (one descriptor per block, issued by wave 0), then all
// waves consume it with 16B ds loads. B per-lane loads are contiguous v8bf.
__global__ void __launch_bounds__(256)
k_proj(const __bf16* __restrict__ Cbf, const __bf16* __restrict__ Vt,
       float* __restrict__ proj) {
  __shared__ __align__(16) __bf16 shA[16 * TD_];   // 8 KB: 16 edge rows x 256
  int lane = threadIdx.x & 31;
  int wv = threadIdx.x >> 5;
  int hi = lane >> 4, lm = lane & 15;
  long e0 = (long)blockIdx.x * 16;

  if (threadIdx.x < 32) {
    // Tensor DMA descriptor (CDNA5 ISA):
    // 2-D tile: tile_dim0=256 (contiguous bf16), tile_dim1=16 rows,
    // tensor dims == tile dims (tile fully in-bounds), dim0 stride = 256.
    unsigned long long gaddr =
        (unsigned long long)(const void*)(Cbf + (size_t)e0 * TD_);
    unsigned int laddr =
        (unsigned int)(unsigned long long)(const void*)&shA[0]; // LDS byte offset
    u32x4 g0;
    g0[0] = 1u;                                        // count=1 (valid user D#)
    g0[1] = laddr;                                     // lds_addr [63:32]
    g0[2] = (unsigned int)(gaddr & 0xffffffffu);       // global_addr [95:64]
    g0[3] = (unsigned int)((gaddr >> 32) & 0x01ffffffu)// global_addr [120:96]
            | (2u << 30);                              // type=2 ("image")
    i32x8 g1;
    g1[0] = 1 << 16;                  // workgroup_mask=0, data_size=1 (2 bytes)
    g1[1] = TD_ << 16;                // tensor_dim0[15:0]=256 at [63:48]
    g1[2] = 16 << 16;                 // tensor_dim0 hi=0 | tensor_dim1[15:0]=16
    g1[3] = TD_ << 16;                // tensor_dim1 hi=0 | tile_dim0=256
    g1[4] = 16;                       // tile_dim1=16, tile_dim2=0
    g1[5] = TD_;                      // tensor_dim0_stride[31:0] = 256
    g1[6] = 0;                        // stride0 hi | tensor_dim1_stride lo
    g1[7] = 0;                        // tensor_dim1_stride hi
    i32x4 gz4 = {0, 0, 0, 0};         // groups 2/3 unused (2-D tensor)
    i32x8 gz8 = {0, 0, 0, 0, 0, 0, 0, 0};
    __builtin_amdgcn_tensor_load_to_lds(g0, g1, gz4, gz4, gz8, 0);
    __builtin_amdgcn_s_wait_tensorcnt(0);              // TENSORcnt == 0
  }
  __syncthreads();

  const __bf16* Arow = &shA[lm * TD_];
  const __bf16* Brow = Vt + (size_t)(wv * 16 + lm) * TD_;
  v8f acc = {};
#pragma unroll
  for (int k0 = 0; k0 < TD_; k0 += 32) {
    v16bf a, b;
    *((v8bf*)&a)       = *(const v8bf*)(Arow + k0 + 8 * hi);
    *(((v8bf*)&a) + 1) = *(const v8bf*)(Arow + k0 + 16 + 8 * hi);
    *((v8bf*)&b)       = *(const v8bf*)(Brow + k0 + 8 * hi);
    *(((v8bf*)&b) + 1) = *(const v8bf*)(Brow + k0 + 16 + 8 * hi);
    acc = __builtin_amdgcn_wmma_f32_16x16x32_bf16(false, a, false, b,
                                                  (short)0, acc, false, false);
  }
#pragma unroll
  for (int v = 0; v < 8; ++v)
    proj[(size_t)(e0 + v + 8 * hi) * D_ + wv * 16 + lm] = acc[v];
}

__global__ void k_projnorm(float* __restrict__ proj,
                           const float* __restrict__ ake0, const float* __restrict__ ake1,
                           float* __restrict__ elog_e) {
  __shared__ float red[128];
  long e = blockIdx.x;
  int d = threadIdx.x;
  float v = proj[(size_t)e * D_ + d];
  if (isinf(v)) v = 0.f;
  float ss = blockReduceSum128(v * v, red, d);
  float vn = v / fmaxf(sqrtf(ss), 1e-12f);
  proj[(size_t)e * D_ + d] = vn;
  float l0 = blockReduceSum128(vn * ake0[d], red, d);
  if (d == 0) elog_e[e] = l0;
  float l1 = blockReduceSum128(vn * ake1[d], red, d);
  if (d == 0) elog_e[E_ + e] = l1;
}

__global__ void k_elog(const float* __restrict__ r_val, const int* __restrict__ relid,
                       const float* __restrict__ rel_lg, float* __restrict__ elog_r) {
  int e = blockIdx.x * 256 + threadIdx.x;
  if (e >= E_) return;
  float rv = r_val[e];
  float sg = (rv > 0.f) ? 1.f : ((rv < 0.f) ? -1.f : 0.f);
  int r = relid[e];
  elog_r[e]      = sg * rel_lg[r];
  elog_r[E_ + e] = sg * rel_lg[R_ + r];
}

__global__ void k_rowptr(const int* __restrict__ adj0, int* __restrict__ row_ptr) {
  int n = blockIdx.x * 256 + threadIdx.x;
  if (n > N_) return;
  int lo = 0, hi = E_;
  while (lo < hi) { int mid = (lo + hi) >> 1; if (adj0[mid] < n) lo = mid + 1; else hi = mid; }
  row_ptr[n] = lo;
}

// ------------------------- per-layer aggregation ----------------------------
// One wave per node: segment softmax (3 passes), Householder reflections,
// weighted segment sum, tanh. Each lane owns 4 of 128 dims.
__global__ void __launch_bounds__(256)
k_agg(const float* __restrict__ feats, const float* __restrict__ rel_nrm,
      const float* __restrict__ proj, const int* __restrict__ adj1,
      const int* __restrict__ relid, const float* __restrict__ r_val,
      const float* __restrict__ lrel, const float* __restrict__ lent,
      const int* __restrict__ row_ptr, float* __restrict__ feats_next,
      float* __restrict__ out, int outBlk) {
  int gwave = (blockIdx.x * blockDim.x + threadIdx.x) >> 5;
  int lane = threadIdx.x & 31;
  if (gwave >= N_) return;
  int n = gwave;
  int eb = row_ptr[n], ee = row_ptr[n + 1];

  // pass 1: segment maxes
  float m1 = -3.402823e38f, m2 = -3.402823e38f;
  for (int e = eb + lane; e < ee; e += 32) { m1 = fmaxf(m1, lrel[e]); m2 = fmaxf(m2, lent[e]); }
#pragma unroll
  for (int o = 16; o > 0; o >>= 1) {
    m1 = fmaxf(m1, __shfl_xor(m1, o, 32));
    m2 = fmaxf(m2, __shfl_xor(m2, o, 32));
  }
  // pass 2: segment exp-sums
  float s1 = 0.f, s2 = 0.f;
  for (int e = eb + lane; e < ee; e += 32) { s1 += expf(lrel[e] - m1); s2 += expf(lent[e] - m2); }
#pragma unroll
  for (int o = 16; o > 0; o >>= 1) { s1 += __shfl_xor(s1, o, 32); s2 += __shfl_xor(s2, o, 32); }
  float inv1 = (s1 > 0.f) ? 1.f / s1 : 0.f;
  float inv2 = (s2 > 0.f) ? 1.f / s2 : 0.f;

  // pass 3: reflections + weighted sum
  float a0 = 0.f, a1 = 0.f, a2 = 0.f, a3 = 0.f;
  for (int e = eb; e < ee; ++e) {
    int nb = adj1[e];
    if (e + 1 < ee) __builtin_prefetch(feats + (size_t)adj1[e + 1] * D_, 0, 1);
    int r = relid[e];
    float z = (r_val[e] != 0.f) ? 1.f : 0.f;   // r_val==0 -> tri_rel==0 -> no reflection
    const float* np = feats   + (size_t)nb * D_;
    const float* tp = rel_nrm + (size_t)r * D_;
    const float* cp = proj    + (size_t)e * D_;
    float n0 = np[lane], n1 = np[lane + 32], n2 = np[lane + 64], n3 = np[lane + 96];
    float t0 = tp[lane], t1 = tp[lane + 32], t2 = tp[lane + 64], t3 = tp[lane + 96];
    float c0 = cp[lane], c1 = cp[lane + 32], c2 = cp[lane + 64], c3 = cp[lane + 96];
    float dr = n0 * t0 + n1 * t1 + n2 * t2 + n3 * t3;
    float de = n0 * c0 + n1 * c1 + n2 * c2 + n3 * c3;
#pragma unroll
    for (int o = 16; o > 0; o >>= 1) { dr += __shfl_xor(dr, o, 32); de += __shfl_xor(de, o, 32); }
    float w1 = expf(lrel[e] - m1) * inv1;
    float w2 = 0.1f * expf(lent[e] - m2) * inv2;
    float k1 = 2.f * dr * z;
    float k2 = 2.f * de;
    a0 += w1 * (n0 - k1 * t0) + w2 * (n0 - k2 * c0);
    a1 += w1 * (n1 - k1 * t1) + w2 * (n1 - k2 * c1);
    a2 += w1 * (n2 - k1 * t2) + w2 * (n2 - k2 * c2);
    a3 += w1 * (n3 - k1 * t3) + w2 * (n3 - k2 * c3);
  }
  a0 = tanhf(a0); a1 = tanhf(a1); a2 = tanhf(a2); a3 = tanhf(a3);
  float* fp = feats_next + (size_t)n * D_;
  fp[lane] = a0; fp[lane + 32] = a1; fp[lane + 64] = a2; fp[lane + 96] = a3;
  float* op = out + (size_t)n * 384 + outBlk;
  op[lane] = a0; op[lane + 32] = a1; op[lane + 64] = a2; op[lane + 96] = a3;
}

// ------------------------- host orchestration -------------------------------

extern "C" void kernel_launch(void* const* d_in, const int* in_sizes, int n_in,
                              void* d_out, int out_size, void* d_ws, size_t ws_size,
                              hipStream_t stream) {
  const float* features = (const float*)d_in[0];  // [N,D]
  const float* rel_emb  = (const float*)d_in[1];  // [R,D]
  const int*   adj      = (const int*)d_in[2];    // [2,E]
  const int*   r_index  = (const int*)d_in[3];    // [2,E] (row0 = arange(E))
  const float* r_val    = (const float*)d_in[4];  // [E]
  const float* fc       = (const float*)d_in[5];  // [N,D]
  const float* ak0      = (const float*)d_in[6];  // [D,1]
  const float* ake0     = (const float*)d_in[7];
  const float* ak1      = (const float*)d_in[8];
  const float* ake1     = (const float*)d_in[9];
  const int* adj0  = adj;
  const int* adj1  = adj + E_;
  const int* relid = r_index + E_;
  float* out = (float*)d_out;

  char* wsp = (char*)d_ws;
  auto carve = [&](size_t bytes) -> void* {
    void* p = (void*)wsp;
    wsp += (bytes + 255) & ~(size_t)255;
    return p;
  };
  float*  feats   = (float*) carve((size_t)N_ * D_ * 4);
  float*  feats2  = (float*) carve((size_t)N_ * D_ * 4);
  float*  rel_nrm = (float*) carve((size_t)R_ * D_ * 4);
  float*  rel_lg  = (float*) carve((size_t)2 * R_ * 4);
  float*  mu      = (float*) carve((size_t)TD_ * 4);
  float*  G       = (float*) carve((size_t)TD_ * TD_ * 4);
  float*  Vw      = (float*) carve((size_t)TD_ * TD_ * 4);
  __bf16* Vt      = (__bf16*)carve((size_t)D_ * TD_ * 2);
  __bf16* Cbf     = (__bf16*)carve((size_t)E_ * TD_ * 2);   // 204.8 MB
  float*  proj    = (float*) carve((size_t)E_ * D_ * 4);    // 204.8 MB
  float*  elog_r  = (float*) carve((size_t)2 * E_ * 4);
  float*  elog_e  = (float*) carve((size_t)2 * E_ * 4);
  int*    row_ptr = (int*)   carve((size_t)(N_ + 1) * 4);

  // ---- layer-invariant phase ----
  k_tanh   <<<(N_ * D_ + 255) / 256, 256, 0, stream>>>(features, feats, out);
  k_relnorm<<<R_, 128, 0, stream>>>(rel_emb, ak0, ak1, rel_nrm, rel_lg);
  k_zero   <<<1, 256, 0, stream>>>(mu, TD_);
  k_zero   <<<(TD_ * TD_) / 256, 256, 0, stream>>>(G, TD_ * TD_);
  k_mu     <<<(E_ + 1023) / 1024, 256, 0, stream>>>(adj0, adj1, fc, mu);
  k_buildC <<<E_, 256, 0, stream>>>(adj0, adj1, fc, Cbf);
  k_gram   <<<dim3(256, (E_ + 4095) / 4096), 32, 0, stream>>>(Cbf, G);
  k_gfix   <<<(TD_ * TD_) / 256, 256, 0, stream>>>(G, mu);
  k_initV  <<<(TD_ * TD_) / 256, 256, 0, stream>>>(Vw);
  k_jacobi <<<1, 256, 0, stream>>>(G, Vw);
  k_prepV  <<<1, 256, 0, stream>>>(G, Vw, Vt);
  k_proj   <<<E_ / 16, 256, 0, stream>>>(Cbf, Vt, proj);
  k_projnorm<<<E_, 128, 0, stream>>>(proj, ake0, ake1, elog_e);
  k_elog   <<<(E_ + 255) / 256, 256, 0, stream>>>(r_val, relid, rel_lg, elog_r);
  k_rowptr <<<(N_ + 1 + 255) / 256, 256, 0, stream>>>(adj0, row_ptr);

  // ---- layer 1: feats -> feats2, out[:,128:256] ----
  k_agg<<<N_ / 8, 256, 0, stream>>>(feats, rel_nrm, proj, adj1, relid, r_val,
                                    elog_r, elog_e, row_ptr, feats2, out, 128);
  // ---- layer 2: feats2 -> feats, out[:,256:384] ----
  k_agg<<<N_ / 8, 256, 0, stream>>>(feats2, rel_nrm, proj, adj1, relid, r_val,
                                    elog_r + E_, elog_e + E_, row_ptr, feats, out, 256);
}
